// QuantizedAntFinal_23888608100972
// MI455X (gfx1250) — 8-bit matrix operand order repaired and checked by reading, compiled, not run on hardware
//
#include <hip/hip_runtime.h>
#include <stdint.h>
#include <stddef.h>


typedef __attribute__((ext_vector_type(8))) int v8i;

static constexpr int BM   = 8192;
static constexpr int KD   = 2048;
static constexpr int ND   = 8192;
static constexpr int KT   = KD / 64;
static constexpr int ATILE = 1024;
static constexpr float QMAXF = 127.0f;


__global__ void k_init(unsigned* __restrict__ u) {
    if (threadIdx.x == 0) { u[0] = 0u; u[1] = 0u; }
}

__global__ __launch_bounds__(256)
void k_absmax_x(const float4* __restrict__ x4, int n4, unsigned* __restrict__ out_bits) {
    __shared__ float red[256];
    float m = 0.f;
    for (int i = blockIdx.x * blockDim.x + threadIdx.x; i < n4; i += gridDim.x * blockDim.x) {
        float4 v = x4[i];
        m = fmaxf(m, fmaxf(fmaxf(fabsf(v.x), fabsf(v.y)), fmaxf(fabsf(v.z), fabsf(v.w))));
    }
    red[threadIdx.x] = m;
    __syncthreads();
    for (int s = 128; s > 0; s >>= 1) {
        if (threadIdx.x < s) red[threadIdx.x] = fmaxf(red[threadIdx.x], red[threadIdx.x + s]);
        __syncthreads();
    }
    if (threadIdx.x == 0) atomicMax(out_bits, __float_as_uint(red[0]));
}

__global__ __launch_bounds__(256)
void k_rowmax_w(const float* __restrict__ w, float* __restrict__ s_w) {
    __shared__ float red[256];
    const int row = blockIdx.x;
    const float4* r4 = (const float4*)(w + (size_t)row * KD);
    float m = 0.f;
    for (int i = threadIdx.x; i < KD / 4; i += 256) {
        float4 v = r4[i];
        m = fmaxf(m, fmaxf(fmaxf(fabsf(v.x), fabsf(v.y)), fmaxf(fabsf(v.z), fabsf(v.w))));
    }
    red[threadIdx.x] = m;
    __syncthreads();
    for (int s = 128; s > 0; s >>= 1) {
        if (threadIdx.x < s) red[threadIdx.x] = fmaxf(red[threadIdx.x], red[threadIdx.x + s]);
        __syncthreads();
    }
    if (threadIdx.x == 0) s_w[row] = red[0] * (1.0f / QMAXF);
}

__device__ __forceinline__ char q8(float v, float inv) {
    float r = __builtin_rintf(v * inv);
    r = fminf(fmaxf(r, -QMAXF), QMAXF);
    return (char)(int)r;
}

__global__ __launch_bounds__(256)
void k_quant_x_swz(const float4* __restrict__ x4, const unsigned* __restrict__ maxbits,
                   char* __restrict__ xq, int n16) {
    const int i = blockIdx.x * blockDim.x + threadIdx.x;
    if (i >= n16) return;
    const float mx = __uint_as_float(*maxbits);
    const float inv = mx > 0.f ? (QMAXF / mx) : 0.f;
    union { int2 d[2]; char c[16]; } u;
#pragma unroll
    for (int j = 0; j < 4; ++j) {
        float4 v = x4[i * 4 + j];
        u.c[j * 4 + 0] = q8(v.x, inv);
        u.c[j * 4 + 1] = q8(v.y, inv);
        u.c[j * 4 + 2] = q8(v.z, inv);
        u.c[j * 4 + 3] = q8(v.w, inv);
    }
    const int row    = i / (KD / 16);
    const int kstart = (i % (KD / 16)) * 16;
    const int tm  = row >> 4;
    const int l16 = row & 15;
    const int tk  = kstart >> 6;
    const int c   = (kstart >> 4) & 3;
    char* tile = xq + ((size_t)tm * KT + tk) * ATILE + (c >> 1) * 16;
    *(int2*)(tile + ((c & 1) * 16 + l16) * 32)     = u.d[0];
    *(int2*)(tile + ((c & 1) * 16 + l16) * 32 + 8) = u.d[1];
}

__global__ __launch_bounds__(256)
void k_quant_w(const float4* __restrict__ w4, const float* __restrict__ s_w,
               int4* __restrict__ wq4, int n16) {
    const int i = blockIdx.x * blockDim.x + threadIdx.x;
    if (i >= n16) return;
    const int row = i / (KD / 16);
    const float sw = s_w[row];
    const float inv = sw > 0.f ? (1.0f / sw) : 0.f;
    union { int4 v; char c[16]; } u;
#pragma unroll
    for (int j = 0; j < 4; ++j) {
        float4 v = w4[i * 4 + j];
        u.c[j * 4 + 0] = q8(v.x, inv);
        u.c[j * 4 + 1] = q8(v.y, inv);
        u.c[j * 4 + 2] = q8(v.z, inv);
        u.c[j * 4 + 3] = q8(v.w, inv);
    }
    wq4[i] = u.v;
}

__global__ __launch_bounds__(256)
void k_bias(const float* __restrict__ bias, const float* __restrict__ s_w,
            const unsigned* __restrict__ maxbits, float* __restrict__ b_f) {
    const int j = blockIdx.x * blockDim.x + threadIdx.x;
    if (j >= ND) return;
    const float s_a = __uint_as_float(*maxbits) * (1.0f / QMAXF);
    const float sb = s_w[j] * s_a;
    b_f[j] = sb != 0.f ? __builtin_rintf(bias[j] / sb) : 0.f;
}

__global__ __launch_bounds__(256)
void k_gemm_q8(const char* __restrict__ Asw, const char* __restrict__ Wq,
               const float* __restrict__ s_w, const float* __restrict__ b_f,
               const unsigned* __restrict__ ubits, float* __restrict__ out,
               unsigned* __restrict__ maxout_bits) {
    __shared__ float red[256];
    const int lane = threadIdx.x & 31;
    const int wave = threadIdx.x >> 5;
    const int wm = wave & 1;
    const int wn = wave >> 1;
    const int m0 = blockIdx.y * 128 + wm * 64;
    const int n0 = blockIdx.x * 128 + wn * 32;
    const int half = lane >> 4;
    const int l16  = lane & 15;
    const float s_a = __uint_as_float(ubits[0]) * (1.0f / QMAXF);

    v8i acc[4][2];
#pragma unroll
    for (int mt = 0; mt < 4; ++mt)
#pragma unroll
        for (int nt = 0; nt < 2; ++nt) acc[mt][nt] = (v8i)0;

    const char* aP = Asw + (size_t)(m0 >> 4) * KT * ATILE + lane * 32;
    const char* bP = Wq + (size_t)(n0 + l16) * KD + half * 16;

    for (int kt = 0; kt < KT; ++kt) {
        v8i a[4], b[2];
#pragma unroll
        for (int mt = 0; mt < 4; ++mt) {
            union { v8i v; int4 q[2]; } ua;
            const char* p = aP + (size_t)mt * (KT * ATILE) + (size_t)kt * ATILE;
            ua.q[0] = *(const int4*)(p);
            ua.q[1] = *(const int4*)(p + 16);
            a[mt] = ua.v;
        }
#pragma unroll
        for (int nt = 0; nt < 2; ++nt) {
            union { v8i v; int4 q[2]; } ub;
            const char* p = bP + (size_t)nt * (16 * KD) + (size_t)kt * 64;
            ub.q[0] = *(const int4*)(p);
            ub.q[1] = *(const int4*)(p + 32);
            b[nt] = ub.v;
        }
#pragma unroll
        for (int mt = 0; mt < 4; ++mt)
#pragma unroll
            for (int nt = 0; nt < 2; ++nt)
                acc[mt][nt] = __builtin_amdgcn_wmma_i32_16x16x64_iu8(
                     true, a[mt],  true, b[nt],
                    acc[mt][nt],  false,  false);
    }

    float lmax = 0.f;
#pragma unroll
    for (int nt = 0; nt < 2; ++nt) {
        const int col = n0 + nt * 16 + l16;
        const float sb = s_w[col] * s_a;
        const float bf = b_f[col];
#pragma unroll
        for (int mt = 0; mt < 4; ++mt) {
            const int rowbase = m0 + mt * 16 + half * 8;
#pragma unroll
            for (int r = 0; r < 8; ++r) {
                float v = ((float)acc[mt][nt][r] + bf) * sb;
                out[(size_t)(rowbase + r) * ND + col] = v;
                lmax = fmaxf(lmax, fabsf(v));
            }
        }
    }
    red[threadIdx.x] = lmax;
    __syncthreads();
    for (int s = 128; s > 0; s >>= 1) {
        if (threadIdx.x < s) red[threadIdx.x] = fmaxf(red[threadIdx.x], red[threadIdx.x + s]);
        __syncthreads();
    }
    if (threadIdx.x == 0) atomicMax(maxout_bits, __float_as_uint(red[0]));
}

__global__ __launch_bounds__(256)
void k_requant(float4* __restrict__ out4, int n4, const unsigned* __restrict__ maxbits) {
    const float mo = __uint_as_float(*maxbits);
    const float s_o = mo * (1.0f / QMAXF);
    const float inv = mo > 0.f ? (QMAXF / mo) : 0.f;
    for (int i = blockIdx.x * blockDim.x + threadIdx.x; i < n4; i += gridDim.x * blockDim.x) {
        float4 v = out4[i];
        v.x = fminf(fmaxf(__builtin_rintf(v.x * inv), -QMAXF), QMAXF) * s_o;
        v.y = fminf(fmaxf(__builtin_rintf(v.y * inv), -QMAXF), QMAXF) * s_o;
        v.z = fminf(fmaxf(__builtin_rintf(v.z * inv), -QMAXF), QMAXF) * s_o;
        v.w = fminf(fmaxf(__builtin_rintf(v.w * inv), -QMAXF), QMAXF) * s_o;
        out4[i] = v;
    }
}

extern "C" void kernel_launch(void* const* d_in, const int* in_sizes, int n_in,
                              void* d_out, int out_size, void* d_ws, size_t ws_size,
                              hipStream_t stream) {
    const float* x    = (const float*)d_in[0];
    const float* w    = (const float*)d_in[1];
    const float* bias = (const float*)d_in[2];
    float* out = (float*)d_out;

    unsigned* ubits = (unsigned*)d_ws;
    float* s_w = (float*)d_ws + 16;
    float* b_f = s_w + ND;
    char*  xq  = (char*)(b_f + ND);
    char*  wq  = xq + (size_t)BM * KD;

    k_init<<<1, 32, 0, stream>>>(ubits);
    k_absmax_x<<<2048, 256, 0, stream>>>((const float4*)x, BM * KD / 4, ubits);
    k_rowmax_w<<<ND, 256, 0, stream>>>(w, s_w);
    k_quant_x_swz<<<(BM * KD / 16 + 255) / 256, 256, 0, stream>>>(
        (const float4*)x, ubits, xq, BM * KD / 16);
    k_quant_w<<<(ND * KD / 16 + 255) / 256, 256, 0, stream>>>(
        (const float4*)w, s_w, (int4*)wq, ND * KD / 16);
    k_bias<<<(ND + 255) / 256, 256, 0, stream>>>(bias, s_w, ubits, b_f);

    dim3 grid(ND / 128, BM / 128);
    k_gemm_q8<<<grid, 256, 0, stream>>>(xq, wq, s_w, b_f, ubits, out, ubits + 1);

    k_requant<<<8192, 256, 0, stream>>>((float4*)out, (int)((size_t)BM * ND / 4), ubits + 1);
}
